// CBSModel_20229295964422
// MI455X (gfx1250) — compile-verified, matrix operand order checked by reading, not run on hardware
//
#include <hip/hip_runtime.h>

// ---------------- problem constants (match reference) ----------------
#define N_NODES   50000
#define N_EDGES   800000
#define FEAT      64
#define N_GRAPHS  512
#define N_CONVS   5

// ---------------- WMMA vector types (CDNA5, wave32) ----------------
typedef __attribute__((ext_vector_type(16))) __bf16 v16bf;
typedef __attribute__((ext_vector_type(8)))  float  v8f;

struct B32 { uint4 a, b; };   // 32-byte carrier for bit_cast into v16bf

__device__ __forceinline__ __bf16 f2bf(float f) {
  // round-to-nearest-even f32 -> bf16
  unsigned u = __builtin_bit_cast(unsigned, f);
  u += 0x7FFFu + ((u >> 16) & 1u);
  unsigned short s = (unsigned short)(u >> 16);
  return __builtin_bit_cast(__bf16, s);
}

// ---------------- small utility kernels ----------------
__global__ void zero_f32(float* __restrict__ p, int n) {
  int i = blockIdx.x * blockDim.x + threadIdx.x;
  if (i < n) p[i] = 0.0f;
}

__global__ void cvt_bf16(const float* __restrict__ in, __bf16* __restrict__ out, int n) {
  int i = blockIdx.x * blockDim.x + threadIdx.x;
  if (i < n) out[i] = f2bf(in[i]);
}

// deg[d] += 1 for each edge (self-loop +1 folded into dinv kernel)
__global__ void deg_accum(const int* __restrict__ dst, float* __restrict__ deg) {
  int i = blockIdx.x * blockDim.x + threadIdx.x;
  if (i < N_EDGES) unsafeAtomicAdd(&deg[dst[i]], 1.0f);
}

__global__ void dinv_k(const float* __restrict__ deg, float* __restrict__ dinv) {
  int i = blockIdx.x * blockDim.x + threadIdx.x;
  if (i < N_NODES) dinv[i] = rsqrtf(deg[i] + 1.0f);
}

// ---------------- weight packing into WMMA B-fragment lane order ----------------
// wpack layout: [layer(6)][kstep(2)][ntile(4)][lane(32)][elem(16)] bf16
// B fragment (16-bit B 32x16, mirrors A layout with N on lanes):
//   lane = 16*kh + c ; elem i<8  -> K = ks*32 + kh*8 + i
//                      elem i>=8 -> K = ks*32 + 16 + kh*8 + (i-8)
//   column n = nt*16 + c ; source W[K*64 + n] (row-major [K][N])
__global__ void pack_weights(const float* __restrict__ convw,
                             const float* __restrict__ w1,
                             __bf16* __restrict__ wpack) {
  int idx = blockIdx.x * blockDim.x + threadIdx.x;
  if (idx >= 6 * 2 * 4 * 32) return;
  int layer = idx >> 8;
  int rem   = idx & 255;
  int ks    = rem >> 7;
  int rem2  = rem & 127;
  int nt    = rem2 >> 5;
  int lane  = rem2 & 31;
  int kh = lane >> 4, c = lane & 15;
  const float* W = (layer < N_CONVS) ? (convw + layer * FEAT * FEAT) : w1;
  __bf16* d = wpack + ((((layer * 2 + ks) * 4 + nt) * 32) + lane) * 16;
#pragma unroll
  for (int i = 0; i < 16; ++i) {
    int K = ks * 32 + ((i < 8) ? (kh * 8 + i) : (16 + kh * 8 + (i - 8)));
    d[i] = f2bf(W[K * FEAT + nt * 16 + c]);
  }
}

// ---------------- WMMA GEMM: out[M x 64] = A[M x 64] (bf16) @ Wpack (bf16), f32 out ----
// One wave -> one 16x64 row tile. ntiles = M/16 (M divisible by 16).
__global__ void gemm16x64_bf16(const __bf16* __restrict__ A,
                               const __bf16* __restrict__ Wp,
                               float* __restrict__ out, int ntiles) {
  int lane = threadIdx.x & 31;
  int wave = threadIdx.x >> 5;
  int tile = blockIdx.x * 8 + wave;
  if (tile >= ntiles) return;              // wave-uniform: EXEC stays all-ones for WMMA
  const int rowBase = tile * 16;
  const int r = lane & 15, kh = lane >> 4;
  const __bf16* arow = A + (size_t)(rowBase + r) * FEAT + kh * 8;

  v8f acc[4] = {v8f{}, v8f{}, v8f{}, v8f{}};
#pragma unroll
  for (int ks = 0; ks < 2; ++ks) {
    // A fragment: 16-bit A 16x32 layout -> two 16B chunks at +0 and +16 elems
    const __bf16* ap = arow + ks * 32;
    B32 ta;
    ta.a = *(const uint4*)(ap);
    ta.b = *(const uint4*)(ap + 16);
    v16bf a = __builtin_bit_cast(v16bf, ta);
#pragma unroll
    for (int nt = 0; nt < 4; ++nt) {
      const __bf16* bp = Wp + (((ks * 4 + nt) * 32) + lane) * 16;  // contiguous 32B/lane
      B32 tb;
      tb.a = *(const uint4*)(bp);
      tb.b = *(const uint4*)(bp + 8);
      v16bf b = __builtin_bit_cast(v16bf, tb);
      acc[nt] = __builtin_amdgcn_wmma_f32_16x16x32_bf16(
          /*neg_a=*/false, a, /*neg_b=*/false, b,
          /*c_mod=*/(short)0, acc[nt], /*reuse_a=*/false, /*reuse_b=*/false);
    }
  }
  // C/D layout: VGPR v, lanes 0-15 -> M=v, N=lane; lanes 16-31 -> M=8+v, N=lane-16
  const int r0 = (lane < 16) ? 0 : 8;
  const int c  = lane & 15;
#pragma unroll
  for (int nt = 0; nt < 4; ++nt) {
#pragma unroll
    for (int v = 0; v < 8; ++v) {
      out[(size_t)(rowBase + r0 + v) * FEAT + nt * 16 + c] = acc[nt][v];
    }
  }
}

// ---------------- edge gather/scale/scatter-add (L2-resident, native f32 atomics) ----
__global__ void edge_scatter(const int* __restrict__ src, const int* __restrict__ dst,
                             const float* __restrict__ dinv, const float* __restrict__ hW,
                             float* __restrict__ agg) {
  int t = blockIdx.x * blockDim.x + threadIdx.x;
  int e = t >> 5;
  if (e >= N_EDGES) return;
  int lane = t & 31;
  if (lane == 0) __builtin_prefetch(src + e + 2048, 0, 1);   // global_prefetch_b8
  int s = src[e], d = dst[e];
  float nrm = dinv[s] * dinv[d];
  const float2 v = *(const float2*)(hW + (size_t)s * FEAT + lane * 2);
  float* ap = agg + (size_t)d * FEAT + lane * 2;
  unsafeAtomicAdd(ap,     v.x * nrm);
  unsafeAtomicAdd(ap + 1, v.y * nrm);
}

// ---------------- agg + self-loop + bias (+ReLU), emit f32 and bf16 ----------------
__global__ void combine_k(const float* __restrict__ agg, const float* __restrict__ hW,
                          const float* __restrict__ dinv, const float* __restrict__ bias,
                          float* __restrict__ hout, __bf16* __restrict__ hbf, int doRelu) {
  int i = blockIdx.x * blockDim.x + threadIdx.x;
  if (i >= N_NODES * FEAT) return;
  int n = i >> 6, f = i & 63;
  float di = dinv[n];
  float v = agg[i] + hW[i] * di * di + bias[f];
  if (doRelu) v = fmaxf(v, 0.0f);
  hout[i] = v;
  hbf[i]  = f2bf(v);
}

// ---------------- global add pool ----------------
__global__ void pool_k(const float* __restrict__ h, const int* __restrict__ batch,
                       float* __restrict__ pooled) {
  int i = blockIdx.x * blockDim.x + threadIdx.x;
  if (i >= N_NODES * FEAT) return;
  int n = i >> 6, f = i & 63;
  unsafeAtomicAdd(&pooled[(size_t)batch[n] * FEAT + f], h[i]);
}

// ---------------- final head: relu(hidden + b1) @ w2 + b2, one wave per graph ------
__global__ void mlp2_k(const float* __restrict__ hidden, const float* __restrict__ b1,
                       const float* __restrict__ w2, const float* __restrict__ b2,
                       float* __restrict__ out) {
  int t = blockIdx.x * blockDim.x + threadIdx.x;
  int lane = t & 31;
  int g = t >> 5;
  if (g >= N_GRAPHS) return;
  float s = 0.0f;
#pragma unroll
  for (int j = 0; j < 2; ++j) {
    int f = lane + j * 32;
    float h = hidden[(size_t)g * FEAT + f] + b1[f];
    h = fmaxf(h, 0.0f);
    s += h * w2[f];
  }
#pragma unroll
  for (int off = 16; off > 0; off >>= 1) s += __shfl_down(s, off, 32);
  if (lane == 0) out[g] = s + b2[0];
}

// ---------------- host-side orchestration ----------------
extern "C" void kernel_launch(void* const* d_in, const int* in_sizes, int n_in,
                              void* d_out, int out_size, void* d_ws, size_t ws_size,
                              hipStream_t stream) {
  const float* x      = (const float*)d_in[0];
  const int*   ei     = (const int*)d_in[1];
  const int*   src    = ei;
  const int*   dst    = ei + N_EDGES;
  const int*   batch  = (const int*)d_in[2];
  const float* conv_w = (const float*)d_in[3];
  const float* conv_b = (const float*)d_in[4];
  const float* w1     = (const float*)d_in[5];
  const float* b1     = (const float*)d_in[6];
  const float* w2     = (const float*)d_in[7];
  const float* b2     = (const float*)d_in[8];
  float* out = (float*)d_out;

  char* wsb = (char*)d_ws;
  size_t off = 0;
  auto alloc = [&](size_t bytes) -> void* {
    void* p = wsb + off;
    off += (bytes + 255) & ~(size_t)255;
    return p;
  };
  float*  deg      = (float*)alloc((size_t)N_NODES * 4);
  float*  dinv     = (float*)alloc((size_t)N_NODES * 4);
  float*  hW       = (float*)alloc((size_t)N_NODES * FEAT * 4);
  float*  agg      = (float*)alloc((size_t)N_NODES * FEAT * 4);
  float*  hbuf     = (float*)alloc((size_t)N_NODES * FEAT * 4);
  __bf16* hbf      = (__bf16*)alloc((size_t)N_NODES * FEAT * 2);
  float*  pooled   = (float*)alloc((size_t)N_GRAPHS * FEAT * 4);
  __bf16* pooledbf = (__bf16*)alloc((size_t)N_GRAPHS * FEAT * 2);
  float*  hidden   = (float*)alloc((size_t)N_GRAPHS * FEAT * 4);
  __bf16* wpack    = (__bf16*)alloc((size_t)6 * 4096 * 2);
  (void)ws_size; (void)in_sizes; (void)n_in; (void)out_size;

  const int B = 256;

  // degree / normalization
  zero_f32 <<<(N_NODES + B - 1) / B, B, 0, stream>>>(deg, N_NODES);
  deg_accum<<<(N_EDGES + B - 1) / B, B, 0, stream>>>(dst, deg);
  dinv_k   <<<(N_NODES + B - 1) / B, B, 0, stream>>>(deg, dinv);

  // pack all dense weights into WMMA B-fragment order (5 convs + w1)
  pack_weights<<<(1536 + B - 1) / B, B, 0, stream>>>(conv_w, w1, wpack);

  // initial activations -> bf16
  cvt_bf16<<<(N_NODES * FEAT + B - 1) / B, B, 0, stream>>>(x, hbf, N_NODES * FEAT);

  const int node_tiles = N_NODES / 16;  // 3125, exact
  for (int layer = 0; layer < N_CONVS; ++layer) {
    gemm16x64_bf16<<<(node_tiles + 7) / 8, B, 0, stream>>>(hbf, wpack + layer * 4096, hW, node_tiles);
    zero_f32<<<(N_NODES * FEAT + B - 1) / B, B, 0, stream>>>(agg, N_NODES * FEAT);
    edge_scatter<<<(N_EDGES * 32) / B, B, 0, stream>>>(src, dst, dinv, hW, agg);
    combine_k<<<(N_NODES * FEAT + B - 1) / B, B, 0, stream>>>(
        agg, hW, dinv, conv_b + layer * FEAT, hbuf, hbf, (layer < N_CONVS - 1) ? 1 : 0);
  }

  // pooling + MLP head
  zero_f32<<<(N_GRAPHS * FEAT + B - 1) / B, B, 0, stream>>>(pooled, N_GRAPHS * FEAT);
  pool_k  <<<(N_NODES * FEAT + B - 1) / B, B, 0, stream>>>(hbuf, batch, pooled);
  cvt_bf16<<<(N_GRAPHS * FEAT + B - 1) / B, B, 0, stream>>>(pooled, pooledbf, N_GRAPHS * FEAT);
  gemm16x64_bf16<<<(32 + 7) / 8, B, 0, stream>>>(pooledbf, wpack + 5 * 4096, hidden, 32);
  mlp2_k  <<<(N_GRAPHS * 32 + B - 1) / B, B, 0, stream>>>(hidden, b1, w2, b2, out);
}
